// Qwen3Attention_44882408243706
// MI455X (gfx1250) — matrix operand order repaired and checked by reading, not compiled, not run on hardware
//
#include <hip/hip_runtime.h>
#include <hip/hip_bf16.h>

typedef __attribute__((ext_vector_type(16))) __bf16 v16bf;
typedef __attribute__((ext_vector_type(8)))  __bf16 v8bf;
typedef __attribute__((ext_vector_type(8)))  float  v8f;
typedef __attribute__((ext_vector_type(4)))  unsigned u32x4;
typedef __attribute__((ext_vector_type(8)))  int      i32x8;
typedef __attribute__((ext_vector_type(4)))  int      i32x4;

#define T_SEQ        4096
#define HIDDEN       2048
#define NUM_HEADS    16
#define NUM_KV_HEADS 8
#define HEAD_DIM     128
#define Q_SIZE       (NUM_HEADS * HEAD_DIM)       // 2048
#define KV_SIZE      (NUM_KV_HEADS * HEAD_DIM)    // 1024
#define QKV_N        (Q_SIZE + 2 * KV_SIZE)       // 4096
#define ATTN_SCALE   0.08838834764831845f         // 128^-0.5
#define RMS_EPS      1e-6f

#if __has_builtin(__builtin_amdgcn_tensor_load_to_lds)
#define HAVE_TDM 1
#else
#define HAVE_TDM 0
#endif

// ---------- helpers ----------------------------------------------------------

static __device__ __forceinline__ __bf16 to_bf16(float f) {
  unsigned u = __builtin_bit_cast(unsigned, f);
  unsigned r = u + 0x7FFFu + ((u >> 16) & 1u);   // round-to-nearest-even
  unsigned short h = (unsigned short)(r >> 16);
  return __builtin_bit_cast(__bf16, h);
}

static __device__ __forceinline__ v16bf cat8(v8bf lo, v8bf hi) {
  return __builtin_shufflevector(lo, hi, 0,1,2,3,4,5,6,7,8,9,10,11,12,13,14,15);
}

static __device__ __forceinline__ v8f wmma_bf16(v16bf a, v16bf b, v8f c) {
  // D(f32 16x16) = A(bf16 16x32) * B(bf16 32x16) + C
  return __builtin_amdgcn_wmma_f32_16x16x32_bf16(
      /*neg_a=*/false, a, /*neg_b=*/false, b,
      /*c_mod=*/(short)0, c, /*reuse_a=*/false, /*reuse_b=*/false);
}

#if HAVE_TDM
// Issue one TDM 2D tile load: global (row-major, 2-byte elems) -> LDS packed.
// D# layout per CDNA5 ISA ch.8: group0 = {count, lds_addr, global_addr, type=2},
// group1 = {data_size=1(2B), tensor dims, tile dims, 48-bit dim0 stride}.
// This toolchain exposes the 6-arg builtin:
//   (u32x4 g0, i32x8 g1, i32x4 g2, i32x4 g3, i32x8 g4, i32 cpol)
static __device__ __forceinline__ void tdm_load_tile_2d(
    unsigned lds_addr, const __bf16* gptr,
    unsigned tile_x, unsigned tile_y,
    unsigned tensor_x, unsigned tensor_y,
    unsigned long long stride_x_elems) {
  unsigned long long ga = (unsigned long long)(const void*)gptr;
  u32x4 g0;
  g0[0] = 1u;                                            // count=1, no gather
  g0[1] = lds_addr;                                      // lds_addr [63:32]
  g0[2] = (unsigned)ga;                                  // global_addr lo
  g0[3] = (unsigned)((ga >> 32) & 0x01FFFFFFull)         // global_addr [56:32]
          | 0x80000000u;                                 // type=2 in [127:126]
  i32x8 g1;
  g1[0] = (int)(1u << 16);                               // data_size=1 (2 bytes)
  g1[1] = (int)((tensor_x & 0xFFFFu) << 16);             // tensor_dim0 lo16
  g1[2] = (int)((tensor_x >> 16) | ((tensor_y & 0xFFFFu) << 16));
  g1[3] = (int)((tensor_y >> 16) | (tile_x << 16));      // tile_dim0
  g1[4] = (int)tile_y;                                   // tile_dim1, tile_dim2=0
  g1[5] = (int)(unsigned)(stride_x_elems & 0xFFFFFFFFull);
  g1[6] = (int)(unsigned)((stride_x_elems >> 32) & 0xFFFFull); // stride1=0
  g1[7] = 0;
  i32x4 z4 = {};
  i32x8 z8 = {};
  __builtin_amdgcn_tensor_load_to_lds(g0, g1, z4, z4, z8, 0);
}
#endif

// ---------- fp32 -> bf16 conversion ------------------------------------------

__global__ __launch_bounds__(256) void cvt_f32_bf16_kernel(
    const float* __restrict__ in, __bf16* __restrict__ out, long long n) {
  long long i = ((long long)blockIdx.x * blockDim.x + threadIdx.x) * 4;
  if (i + 3 < n) {
    float4 f = *(const float4*)(in + i);
    out[i + 0] = to_bf16(f.x);
    out[i + 1] = to_bf16(f.y);
    out[i + 2] = to_bf16(f.z);
    out[i + 3] = to_bf16(f.w);
  }
}

// ---------- bf16 WMMA GEMM:  C[M,N] = A[M,K] * B[N,K]^T  ---------------------
// Block tile 128x128, K step 32. 256 threads = 8 waves in a 4x2 grid,
// each wave owns a 32x64 patch = 2x4 WMMA accumulators. Tiles staged into
// LDS by the Tensor Data Mover (one wave issues, TENSORcnt-tracked).

#define BM 128
#define BN 128
#define BK 32

__global__ __launch_bounds__(256) void gemm_bf16_kernel(
    const __bf16* __restrict__ A,   // [M,K] row-major
    const __bf16* __restrict__ B,   // [N,K] row-major (i.e. B^T of math B)
    float* __restrict__ C,          // [M,N]
    int M, int N, int K) {
  __shared__ __bf16 sA[BM][BK];     // 8 KB
  __shared__ __bf16 sB[BN][BK];     // 8 KB

  const int tid  = threadIdx.x;
  const int wave = tid >> 5;
  const int lane = tid & 31;
  const int half = lane >> 4;       // 0..1
  const int l16  = lane & 15;       // 0..15
  const int wm   = wave >> 1;       // 0..3  (M direction, 32 rows each)
  const int wn   = wave & 1;        // 0..1  (N direction, 64 cols each)
  const int blockM = blockIdx.x * BM;
  const int blockN = blockIdx.y * BN;

  v8f acc[2][4] = {};

  for (int k0 = 0; k0 < K; k0 += BK) {
#if HAVE_TDM
    if (wave == 0) {
      tdm_load_tile_2d((unsigned)(size_t)&sA[0][0],
                       A + (long long)blockM * K + k0,
                       BK, BM, (unsigned)K, (unsigned)BM,
                       (unsigned long long)(unsigned)K);
      tdm_load_tile_2d((unsigned)(size_t)&sB[0][0],
                       B + (long long)blockN * K + k0,
                       BK, BN, (unsigned)K, (unsigned)BN,
                       (unsigned long long)(unsigned)K);
      __builtin_amdgcn_s_wait_tensorcnt(0);
    }
    __syncthreads();
#else
    {
      const int r = tid >> 1;              // 0..127
      const int c = (tid & 1) * 16;        // 0 or 16
      const __bf16* srcA = A + (long long)(blockM + r) * K + k0 + c;
      *(v8bf*)(&sA[r][c + 0]) = *(const v8bf*)(srcA + 0);
      *(v8bf*)(&sA[r][c + 8]) = *(const v8bf*)(srcA + 8);
      const __bf16* srcB = B + (long long)(blockN + r) * K + k0 + c;
      *(v8bf*)(&sB[r][c + 0]) = *(const v8bf*)(srcB + 0);
      *(v8bf*)(&sB[r][c + 8]) = *(const v8bf*)(srcB + 8);
    }
    __syncthreads();
#endif

    // ---- A fragments: lane row = l16, chunks K = 8*half and 16+8*half ----
    v16bf afrag[2], bfrag[4];
#pragma unroll
    for (int i = 0; i < 2; ++i) {
      const __bf16* row = &sA[wm * 32 + i * 16 + l16][0];
      v8bf lo = *(const v8bf*)(row + 8 * half);
      v8bf hi = *(const v8bf*)(row + 16 + 8 * half);
      afrag[i] = cat8(lo, hi);
    }
    // ---- B fragments: lane col = l16, chunks K = 8*half and 16+8*half ----
#pragma unroll
    for (int j = 0; j < 4; ++j) {
      const __bf16* row = &sB[wn * 64 + j * 16 + l16][0];
      v8bf lo = *(const v8bf*)(row + 8 * half);
      v8bf hi = *(const v8bf*)(row + 16 + 8 * half);
      bfrag[j] = cat8(lo, hi);
    }
#pragma unroll
    for (int i = 0; i < 2; ++i)
#pragma unroll
      for (int j = 0; j < 4; ++j)
        acc[i][j] = wmma_bf16(afrag[i], bfrag[j], acc[i][j]);
    __syncthreads();
  }

  // ---- store: C-tile layout (m = r + 8*half, n = l16) ----
#pragma unroll
  for (int i = 0; i < 2; ++i)
#pragma unroll
    for (int j = 0; j < 4; ++j)
#pragma unroll
      for (int r = 0; r < 8; ++r) {
        int m = blockM + wm * 32 + i * 16 + r + 8 * half;
        int n = blockN + wn * 64 + j * 16 + l16;
        C[(long long)m * N + n] = acc[i][j][r];
      }
}

// ---------- fused RMSNorm + RoPE + bf16 pack ---------------------------------
// One wave per (token, head-slot). Slots 0..15 = Q heads, 16..23 = K heads
// (norm + rope), 24..31 = V heads (straight convert).

__global__ __launch_bounds__(256) void rmsrope_kernel(
    const float* __restrict__ qkv,     // [T, 4096]
    const float* __restrict__ qw,      // [128]
    const float* __restrict__ kw,      // [128]
    const int*   __restrict__ pos,     // [T]
    __bf16* __restrict__ Qb,           // [T, 2048]
    __bf16* __restrict__ Kb,           // [T, 1024]
    __bf16* __restrict__ Vb) {         // [T, 1024]
  int gw   = (int)((blockIdx.x * blockDim.x + threadIdx.x) >> 5);
  int lane = threadIdx.x & 31;
  int t    = gw >> 5;          // token
  int slot = gw & 31;          // head-slot in qkv row

  const float* x = qkv + (long long)t * QKV_N + slot * HEAD_DIM;

  if (slot < 24) {
    float x0 = x[lane], x1 = x[lane + 32], x2 = x[lane + 64], x3 = x[lane + 96];
    float ss = x0 * x0 + x1 * x1 + x2 * x2 + x3 * x3;
#pragma unroll
    for (int off = 16; off; off >>= 1) ss += __shfl_xor(ss, off, 32);
    float inv = rsqrtf(ss * (1.0f / 128.0f) + RMS_EPS);
    const float* w = (slot < 16) ? qw : kw;
    float y0 = x0 * inv * w[lane];
    float y1 = x1 * inv * w[lane + 32];
    float y2 = x2 * inv * w[lane + 64];
    float y3 = x3 * inv * w[lane + 96];
    // RoPE: pair (i, i+64) rotated by pos * theta^(-2i/128); i = lane, lane+32
    float p = (float)pos[t];
    float a0 = p * __powf(1.0e6f, -(2.0f * (float)lane) * (1.0f / 128.0f));
    float a1 = p * __powf(1.0e6f, -(2.0f * (float)(lane + 32)) * (1.0f / 128.0f));
    float c0 = __cosf(a0), s0 = __sinf(a0);
    float c1 = __cosf(a1), s1 = __sinf(a1);
    float r0 = y0 * c0 - y2 * s0;
    float r1 = y1 * c1 - y3 * s1;
    float r2 = y2 * c0 + y0 * s0;
    float r3 = y3 * c1 + y1 * s1;
    __bf16* dst = (slot < 16)
        ? (Qb + (long long)t * Q_SIZE + slot * HEAD_DIM)
        : (Kb + (long long)t * KV_SIZE + (slot - 16) * HEAD_DIM);
    dst[lane]      = to_bf16(r0);
    dst[lane + 32] = to_bf16(r1);
    dst[lane + 64] = to_bf16(r2);
    dst[lane + 96] = to_bf16(r3);
  } else {
    __bf16* dst = Vb + (long long)t * KV_SIZE + (slot - 24) * HEAD_DIM;
    dst[lane]      = to_bf16(x[lane]);
    dst[lane + 32] = to_bf16(x[lane + 32]);
    dst[lane + 64] = to_bf16(x[lane + 64]);
    dst[lane + 96] = to_bf16(x[lane + 96]);
  }
}

// ---------- causal flash attention (WMMA, online softmax) --------------------
// Grid: (T/64, NUM_HEADS). 128 threads = 4 waves; each wave owns 16 q rows.

#define KB 32   // key block

__global__ __launch_bounds__(128) void attn_kernel(
    const __bf16* __restrict__ Q,   // [T, 2048]
    const __bf16* __restrict__ Kx,  // [T, 1024]
    const __bf16* __restrict__ Vx,  // [T, 1024]
    __bf16* __restrict__ Ob) {      // [T, 2048]
  __shared__ __bf16 sK[KB][HEAD_DIM];    // 8 KB  (key-major: QK^T B-frags)
  __shared__ __bf16 sVt[HEAD_DIM][KB];   // 8 KB  (d-major:   PV  B-frags)
  __shared__ __bf16 sP[4][16][KB];       // 4 KB  (per-wave P re-striping)

  const int tid  = threadIdx.x;
  const int wave = tid >> 5;
  const int lane = tid & 31;
  const int half = lane >> 4;
  const int l16  = lane & 15;
  const int h    = blockIdx.y;
  const int kvh  = h >> 1;                    // GQA rep=2
  const int qb   = blockIdx.x * 64;
  const int qrow = qb + wave * 16;

  // Q fragments for this wave's 16 rows (A layout), kept resident
  v16bf qfrag[4];
  {
    const __bf16* qp = Q + (long long)(qrow + l16) * Q_SIZE + h * HEAD_DIM;
#pragma unroll
    for (int s = 0; s < 4; ++s) {
      v8bf lo = *(const v8bf*)(qp + s * 32 + 8 * half);
      v8bf hi = *(const v8bf*)(qp + s * 32 + 16 + 8 * half);
      qfrag[s] = cat8(lo, hi);
    }
  }

  v8f o[8] = {};
  float m8[8], l8[8];
#pragma unroll
  for (int r = 0; r < 8; ++r) { m8[r] = -INFINITY; l8[r] = 0.0f; }

  const int kend = qb + 64;   // causal: only key blocks with jb < qb+64
  for (int jb = 0; jb < kend; jb += KB) {
    // ---- stage K block row-major and V block transposed ----
    {
      int key = tid >> 2;                // 0..31
      int d0  = (tid & 3) * 32;          // 0,32,64,96
      const __bf16* ks = Kx + (long long)(jb + key) * KV_SIZE + kvh * HEAD_DIM + d0;
#pragma unroll
      for (int u = 0; u < 4; ++u)
        *(v8bf*)(&sK[key][d0 + u * 8]) = *(const v8bf*)(ks + u * 8);
      const __bf16* vs = Vx + (long long)(jb + key) * KV_SIZE + kvh * HEAD_DIM + d0;
#pragma unroll
      for (int u = 0; u < 32; ++u)
        sVt[d0 + u][key] = vs[u];
    }
    __syncthreads();

    // ---- S = Q K^T for two 16-key subtiles ----
    v8f s0 = {}, s1 = {};
#pragma unroll
    for (int s = 0; s < 4; ++s) {
      v8bf lo0 = *(const v8bf*)(&sK[l16][s * 32 + 8 * half]);
      v8bf hi0 = *(const v8bf*)(&sK[l16][s * 32 + 16 + 8 * half]);
      s0 = wmma_bf16(qfrag[s], cat8(lo0, hi0), s0);
      v8bf lo1 = *(const v8bf*)(&sK[16 + l16][s * 32 + 8 * half]);
      v8bf hi1 = *(const v8bf*)(&sK[16 + l16][s * 32 + 16 + 8 * half]);
      s1 = wmma_bf16(qfrag[s], cat8(lo1, hi1), s1);
    }

    // ---- scale + causal mask (C layout: row = r+8*half, col = l16) ----
#pragma unroll
    for (int r = 0; r < 8; ++r) {
      int qg = qrow + r + 8 * half;
      int k0g = jb + l16, k1g = jb + 16 + l16;
      s0[r] = (k0g <= qg) ? s0[r] * ATTN_SCALE : -INFINITY;
      s1[r] = (k1g <= qg) ? s1[r] * ATTN_SCALE : -INFINITY;
    }

    // ---- online softmax: row stats via 16-lane shfl reductions ----
    float alpha8[8];
#pragma unroll
    for (int r = 0; r < 8; ++r) {
      float mx = fmaxf(s0[r], s1[r]);
#pragma unroll
      for (int off = 8; off; off >>= 1) mx = fmaxf(mx, __shfl_xor(mx, off, 32));
      float mn = fmaxf(m8[r], mx);
      alpha8[r] = __expf(m8[r] - mn);
      m8[r] = mn;
      s0[r] = __expf(s0[r] - mn);
      s1[r] = __expf(s1[r] - mn);
      float sm = s0[r] + s1[r];
#pragma unroll
      for (int off = 8; off; off >>= 1) sm += __shfl_xor(sm, off, 32);
      l8[r] = alpha8[r] * l8[r] + sm;
    }
#pragma unroll
    for (int n = 0; n < 8; ++n)
#pragma unroll
      for (int r = 0; r < 8; ++r) o[n][r] *= alpha8[r];

    // ---- re-stripe P: C layout -> A-fragment layout via per-wave LDS ----
#pragma unroll
    for (int r = 0; r < 8; ++r) {
      sP[wave][r + 8 * half][l16]      = to_bf16(s0[r]);
      sP[wave][r + 8 * half][16 + l16] = to_bf16(s1[r]);
    }
    // same-wave DS ordering guarantees RAW here (no cross-wave sharing)
    v8bf plo = *(const v8bf*)(&sP[wave][l16][8 * half]);
    v8bf phi = *(const v8bf*)(&sP[wave][l16][16 + 8 * half]);
    v16bf pfrag = cat8(plo, phi);

    // ---- O += P @ V (B-frags contiguous from transposed V) ----
#pragma unroll
    for (int n = 0; n < 8; ++n) {
      v8bf lo = *(const v8bf*)(&sVt[n * 16 + l16][8 * half]);
      v8bf hi = *(const v8bf*)(&sVt[n * 16 + l16][16 + 8 * half]);
      o[n] = wmma_bf16(pfrag, cat8(lo, hi), o[n]);
    }
    __syncthreads();
  }

  // ---- normalize and emit bf16 for the O-projection GEMM ----
#pragma unroll
  for (int n = 0; n < 8; ++n)
#pragma unroll
    for (int r = 0; r < 8; ++r) {
      float val = o[n][r] / l8[r];
      Ob[(long long)(qrow + r + 8 * half) * Q_SIZE + h * HEAD_DIM + n * 16 + l16] =
          to_bf16(val);
    }
}

// ---------- launch -----------------------------------------------------------

extern "C" void kernel_launch(void* const* d_in, const int* in_sizes, int n_in,
                              void* d_out, int out_size, void* d_ws, size_t ws_size,
                              hipStream_t stream) {
  (void)in_sizes; (void)n_in; (void)out_size; (void)ws_size;

  const float* hs     = (const float*)d_in[0];   // [4096, 2048]
  const float* qkv_w  = (const float*)d_in[1];   // [4096, 2048]
  const float* q_norm = (const float*)d_in[2];   // [128]
  const float* k_norm = (const float*)d_in[3];   // [128]
  const float* o_w    = (const float*)d_in[4];   // [2048, 2048]
  const int*   posi   = (const int*)d_in[5];     // [4096]
  float*       out    = (float*)d_out;           // [4096, 2048]

  const size_t MB = 1u << 20;
  char* ws = (char*)d_ws;
  __bf16* hsb   = (__bf16*)(ws + 0 * MB);     // 16 MB  hidden bf16
  __bf16* wqkvb = (__bf16*)(ws + 16 * MB);    // 16 MB  qkv_w bf16
  __bf16* wob   = (__bf16*)(ws + 32 * MB);    // 8  MB  o_w bf16
  float*  qkvf  = (float*)(ws + 40 * MB);     // 64 MB  qkv fp32
  __bf16* qb    = (__bf16*)(ws + 104 * MB);   // 16 MB  q bf16 (post norm/rope)
  __bf16* kb    = (__bf16*)(ws + 120 * MB);   // 8  MB  k bf16
  __bf16* vb    = (__bf16*)(ws + 128 * MB);   // 8  MB  v bf16
  __bf16* attnb = (__bf16*)(ws + 40 * MB);    // 16 MB  attention out (reuses qkvf)

  // 1) fp32 -> bf16 conversions
  cvt_f32_bf16_kernel<<<8192, 256, 0, stream>>>(hs, hsb, (long long)T_SEQ * HIDDEN);
  cvt_f32_bf16_kernel<<<8192, 256, 0, stream>>>(qkv_w, wqkvb, (long long)QKV_N * HIDDEN);
  cvt_f32_bf16_kernel<<<4096, 256, 0, stream>>>(o_w, wob, (long long)HIDDEN * Q_SIZE);

  // 2) QKV projection: [4096,2048] x [4096,2048]^T -> [4096,4096] fp32
  gemm_bf16_kernel<<<dim3(T_SEQ / BM, QKV_N / BN), 256, 0, stream>>>(
      hsb, wqkvb, qkvf, T_SEQ, QKV_N, HIDDEN);

  // 3) RMSNorm + RoPE + pack to bf16
  rmsrope_kernel<<<(T_SEQ * 32 * 32) / 256, 256, 0, stream>>>(
      qkvf, q_norm, k_norm, posi, qb, kb, vb);

  // 4) causal flash attention
  attn_kernel<<<dim3(T_SEQ / 64, NUM_HEADS), 128, 0, stream>>>(qb, kb, vb, attnb);

  // 5) O projection: [4096,2048] x [2048,2048]^T -> d_out fp32
  gemm_bf16_kernel<<<dim3(T_SEQ / BM, HIDDEN / BN), 256, 0, stream>>>(
      attnb, wob, out, T_SEQ, HIDDEN, Q_SIZE);
}
